// Autograd4bitQuantLinear_1915555414159
// MI455X (gfx1250) — matrix operand order repaired and checked by reading, not compiled, not run on hardware
//
#include <hip/hip_runtime.h>

typedef __attribute__((ext_vector_type(16))) _Float16 v16h;
typedef __attribute__((ext_vector_type(8)))  float    v8f;
typedef __attribute__((ext_vector_type(2)))  _Float16 h2;

#define IN_F   4096
#define OUT_F  11008
#define GROUPS 32
#define GSIZE  128
#define TOKENS 8192

#define BM 128
#define BN 128
#define BK 64
#define LDSW 72   // halves per LDS row: 64 data + 8 pad (144B stride)

// ---------------------------------------------------------------------------
// Pre-pass 1: x fp32 -> f16, 8-block K-permutation applied: block stores
// (k,k+4),(k+1,k+5),(k+2,k+6),(k+3,k+7) as 8 halves (one uint4 per 8-block).
// ---------------------------------------------------------------------------
__global__ __launch_bounds__(256)
void cvt_x_f16(const float* __restrict__ x, uint4* __restrict__ x16)
{
    const int idx = blockIdx.x * 256 + threadIdx.x;   // TOKENS*IN_F/8 blocks
    const float* p = x + (size_t)idx * 8;
    const float4 lo = *(const float4*)p;
    const float4 hi = *(const float4*)(p + 4);
    union { _Float16 h[8]; uint4 u; } pk;
    pk.h[0] = (_Float16)lo.x;  pk.h[1] = (_Float16)hi.x;
    pk.h[2] = (_Float16)lo.y;  pk.h[3] = (_Float16)hi.y;
    pk.h[4] = (_Float16)lo.z;  pk.h[5] = (_Float16)hi.z;
    pk.h[6] = (_Float16)lo.w;  pk.h[7] = (_Float16)hi.w;
    x16[idx] = pk.u;
}

// ---------------------------------------------------------------------------
// Pre-pass 2: GPTQ 4-bit -> f16 dequant, same K permutation per 8-block.
// Output layout: w16[kblk * OUT_F + n] = 8 halves for k in [kblk*8, kblk*8+8).
// Fully coalesced reads (consecutive n) and writes (consecutive uint4).
// ---------------------------------------------------------------------------
__global__ __launch_bounds__(256)
void deq_w_f16(const int*   __restrict__ qweight,
               const int*   __restrict__ qzeros,
               const float* __restrict__ scales,
               uint4*       __restrict__ w16)
{
    const int n    = blockIdx.x * 256 + threadIdx.x;  // 43*256 = 11008
    const int kblk = blockIdx.y;                      // 512 packed rows
    const int g    = kblk >> 4;                       // (kblk*8)/128

    const unsigned int uq = (unsigned int)qweight[(size_t)kblk * OUT_F + n];
    const int   zq   = qzeros[(size_t)g * (OUT_F / 8) + (n >> 3)];
    const int   znib = (zq >> ((n & 7) * 4)) & 0xF;
    const float sc   = scales[(size_t)g * OUT_F + n];

    h2 sc2, c2;
    sc2[0] = sc2[1] = (_Float16)sc;
    c2[0]  = c2[1]  = (_Float16)(float)(1025 + znib);  // 1024 + (z+1), exact in f16

    union { h2 h[4]; uint4 u; } pk;
    #pragma unroll
    for (int s = 0; s < 4; ++s) {
        union { unsigned int u; h2 h; } t;
        t.u = ((uq >> (4 * s)) & 0x000F000Fu) | 0x64006400u;  // (1024+n_s, 1024+n_{s+4})
        pk.h[s] = (t.h - c2) * sc2;
    }
    w16[(size_t)kblk * OUT_F + n] = pk.u;
}

// ---------------------------------------------------------------------------
// Main GEMM (fast path): both operands pre-converted f16, hot loop is pure
// data movement + WMMA.
// ---------------------------------------------------------------------------
__global__ __launch_bounds__(256)
void gptq4_wmma_gemm_fast(const uint4* __restrict__ x16,   // [m][IN_F/8] permuted blocks
                          const uint4* __restrict__ w16,   // [kblk][n] permuted blocks
                          const float* __restrict__ bias,
                          float*       __restrict__ out)
{
    __shared__ __align__(16) _Float16 ldsA[BM * LDSW];
    __shared__ __align__(16) _Float16 ldsB[BN * LDSW];

    const int tid    = threadIdx.x;
    const int lane   = tid & 31;
    const int wave   = tid >> 5;
    const int laneLo = lane & 15;
    const int laneHi = lane >> 4;

    const int waveM = wave & 1;
    const int waveN = wave >> 1;

    const int n0 = blockIdx.x * BN;
    const int m0 = blockIdx.y * BM;

    v8f acc[4][2];
    #pragma unroll
    for (int mt = 0; mt < 4; ++mt)
        #pragma unroll
        for (int nt = 0; nt < 2; ++nt)
            #pragma unroll
            for (int j = 0; j < 8; ++j)
                acc[mt][nt][j] = 0.0f;

    const int nB = tid & 127;
    const int rb = (tid >> 7) * 4;
    const int ncol = n0 + nB;

    for (int kb = 0; kb < IN_F / BK; ++kb) {
        // ---- global loads into registers (b128, coalesced) ----
        uint4 av[4];
        #pragma unroll
        for (int i = 0; i < 4; ++i) {
            const int lin = tid + i * 256;       // 128 rows x 8 blocks
            const int r   = lin >> 3;
            const int blk = lin & 7;
            av[i] = x16[(size_t)(m0 + r) * (IN_F / 8) + kb * 8 + blk];
        }
        uint4 bw[4];
        #pragma unroll
        for (int j = 0; j < 4; ++j)
            bw[j] = w16[(size_t)(kb * 8 + rb + j) * OUT_F + ncol];

        __syncthreads();

        #pragma unroll
        for (int i = 0; i < 4; ++i) {
            const int lin = tid + i * 256;
            const int r   = lin >> 3;
            const int blk = lin & 7;
            *(uint4*)&ldsA[r * LDSW + blk * 8] = av[i];
        }
        #pragma unroll
        for (int j = 0; j < 4; ++j)
            *(uint4*)&ldsB[nB * LDSW + (rb + j) * 8] = bw[j];

        __syncthreads();

        #pragma unroll
        for (int kk = 0; kk < 2; ++kk) {
            v16h bfrag[2];
            #pragma unroll
            for (int nt = 0; nt < 2; ++nt) {
                const int n  = waveN * 32 + nt * 16 + laneLo;
                const int ka = kk * 32 + laneHi * 8;
                union { v16h v; uint4 u[2]; } f;
                const uint4* p = (const uint4*)&ldsB[n * LDSW + ka];
                f.u[0] = p[0];
                f.u[1] = p[2];
                bfrag[nt] = f.v;
            }
            #pragma unroll
            for (int mt = 0; mt < 4; ++mt) {
                const int r  = waveM * 64 + mt * 16 + laneLo;
                const int ka = kk * 32 + laneHi * 8;
                union { v16h v; uint4 u[2]; } f;
                f.u[0] = *(const uint4*)&ldsA[r * LDSW + ka];
                f.u[1] = *(const uint4*)&ldsA[r * LDSW + ka + 16];
                const v16h afrag = f.v;
                #pragma unroll
                for (int nt = 0; nt < 2; ++nt) {
                    acc[mt][nt] = __builtin_amdgcn_wmma_f32_16x16x32_f16(
                        false, afrag, false, bfrag[nt],
                        (short)0, acc[mt][nt], false, false);
                }
            }
        }
    }

    #pragma unroll
    for (int nt = 0; nt < 2; ++nt) {
        const int n  = n0 + waveN * 32 + nt * 16 + laneLo;
        const float bv = bias[n];
        #pragma unroll
        for (int mt = 0; mt < 4; ++mt) {
            #pragma unroll
            for (int j = 0; j < 8; ++j) {
                const int m = m0 + waveM * 64 + mt * 16 + laneHi * 8 + j;
                out[(size_t)m * OUT_F + n] = acc[mt][nt][j] + bv;
            }
        }
    }
}

// ---------------------------------------------------------------------------
// Fallback: fused dequant GEMM (round-2 kernel) when workspace is too small.
// ---------------------------------------------------------------------------
__global__ __launch_bounds__(256)
void gptq4_wmma_gemm_fused(const float* __restrict__ x,
                           const int*   __restrict__ qweight,
                           const int*   __restrict__ qzeros,
                           const float* __restrict__ scales,
                           const float* __restrict__ bias,
                           float*       __restrict__ out)
{
    __shared__ __align__(16) _Float16 ldsA[BM * LDSW];
    __shared__ __align__(16) _Float16 ldsB[BN * LDSW];

    const int tid    = threadIdx.x;
    const int lane   = tid & 31;
    const int wave   = tid >> 5;
    const int laneLo = lane & 15;
    const int laneHi = lane >> 4;
    const int waveM = wave & 1;
    const int waveN = wave >> 1;
    const int n0 = blockIdx.x * BN;
    const int m0 = blockIdx.y * BM;

    v8f acc[4][2];
    #pragma unroll
    for (int mt = 0; mt < 4; ++mt)
        #pragma unroll
        for (int nt = 0; nt < 2; ++nt)
            #pragma unroll
            for (int j = 0; j < 8; ++j)
                acc[mt][nt][j] = 0.0f;

    const int nB = tid & 127;
    const int rb = (tid >> 7) * 4;
    const int ncol = n0 + nB;
    const int zshift = (ncol & 7) * 4;

    for (int kb = 0; kb < IN_F / BK; ++kb) {
        const int k0 = kb * BK;
        const int g  = k0 / GSIZE;

        float4 alo[4], ahi[4];
        #pragma unroll
        for (int i = 0; i < 4; ++i) {
            const int lin = tid + i * 256;
            const int r   = lin >> 3;
            const int blk = lin & 7;
            const float* p = x + (size_t)(m0 + r) * IN_F + k0 + blk * 8;
            alo[i] = *(const float4*)p;
            ahi[i] = *(const float4*)(p + 4);
        }
        int qv[4];
        #pragma unroll
        for (int j = 0; j < 4; ++j)
            qv[j] = qweight[(size_t)(kb * 8 + rb + j) * OUT_F + ncol];
        const int   zq   = qzeros[(size_t)g * (OUT_F / 8) + (ncol >> 3)];
        const int   znib = (zq >> zshift) & 0xF;
        const float sc   = scales[(size_t)g * OUT_F + ncol];

        h2 sc2, c2;
        sc2[0] = sc2[1] = (_Float16)sc;
        c2[0]  = c2[1]  = (_Float16)(float)(1025 + znib);

        __syncthreads();

        #pragma unroll
        for (int i = 0; i < 4; ++i) {
            const int lin = tid + i * 256;
            const int r   = lin >> 3;
            const int blk = lin & 7;
            union { _Float16 h[8]; uint4 u; } pk;
            pk.h[0] = (_Float16)alo[i].x;  pk.h[1] = (_Float16)ahi[i].x;
            pk.h[2] = (_Float16)alo[i].y;  pk.h[3] = (_Float16)ahi[i].y;
            pk.h[4] = (_Float16)alo[i].z;  pk.h[5] = (_Float16)ahi[i].z;
            pk.h[6] = (_Float16)alo[i].w;  pk.h[7] = (_Float16)ahi[i].w;
            *(uint4*)&ldsA[r * LDSW + blk * 8] = pk.u;
        }
        #pragma unroll
        for (int j = 0; j < 4; ++j) {
            const unsigned int uq = (unsigned int)qv[j];
            union { h2 h[4]; uint4 u; } pk;
            #pragma unroll
            for (int s = 0; s < 4; ++s) {
                union { unsigned int u; h2 h; } t;
                t.u = ((uq >> (4 * s)) & 0x000F000Fu) | 0x64006400u;
                pk.h[s] = (t.h - c2) * sc2;
            }
            *(uint4*)&ldsB[nB * LDSW + (rb + j) * 8] = pk.u;
        }

        __syncthreads();

        #pragma unroll
        for (int kk = 0; kk < 2; ++kk) {
            v16h bfrag[2];
            #pragma unroll
            for (int nt = 0; nt < 2; ++nt) {
                const int n  = waveN * 32 + nt * 16 + laneLo;
                const int ka = kk * 32 + laneHi * 8;
                union { v16h v; uint4 u[2]; } f;
                const uint4* p = (const uint4*)&ldsB[n * LDSW + ka];
                f.u[0] = p[0];
                f.u[1] = p[2];
                bfrag[nt] = f.v;
            }
            #pragma unroll
            for (int mt = 0; mt < 4; ++mt) {
                const int r  = waveM * 64 + mt * 16 + laneLo;
                const int ka = kk * 32 + laneHi * 8;
                union { v16h v; uint4 u[2]; } f;
                f.u[0] = *(const uint4*)&ldsA[r * LDSW + ka];
                f.u[1] = *(const uint4*)&ldsA[r * LDSW + ka + 16];
                const v16h afrag = f.v;
                #pragma unroll
                for (int nt = 0; nt < 2; ++nt) {
                    acc[mt][nt] = __builtin_amdgcn_wmma_f32_16x16x32_f16(
                        false, afrag, false, bfrag[nt],
                        (short)0, acc[mt][nt], false, false);
                }
            }
        }
    }

    #pragma unroll
    for (int nt = 0; nt < 2; ++nt) {
        const int n  = n0 + waveN * 32 + nt * 16 + laneLo;
        const float bv = bias[n];
        #pragma unroll
        for (int mt = 0; mt < 4; ++mt) {
            #pragma unroll
            for (int j = 0; j < 8; ++j) {
                const int m = m0 + waveM * 64 + mt * 16 + laneHi * 8 + j;
                out[(size_t)m * OUT_F + n] = acc[mt][nt][j] + bv;
            }
        }
    }
}

extern "C" void kernel_launch(void* const* d_in, const int* in_sizes, int n_in,
                              void* d_out, int out_size, void* d_ws, size_t ws_size,
                              hipStream_t stream) {
    const float* x       = (const float*)d_in[0];
    const int*   qweight = (const int*)  d_in[1];
    const int*   qzeros  = (const int*)  d_in[2];
    const float* scales  = (const float*)d_in[3];
    // d_in[4] = g_idx: canonical arange(IN_F)/GSIZE, group derived analytically
    const float* bias    = (const float*)d_in[5];
    float*       out     = (float*)d_out;

    const size_t X16_BYTES = (size_t)TOKENS * IN_F * 2;              // 67,108,864
    const size_t W16_BYTES = (size_t)(IN_F / 8) * OUT_F * 16;        // 90,177,536

    dim3 grid(OUT_F / BN, TOKENS / BM);   // 86 x 64
    dim3 block(256);

    if (ws_size >= X16_BYTES + W16_BYTES) {
        uint4* x16 = (uint4*)d_ws;
        uint4* w16 = (uint4*)((char*)d_ws + X16_BYTES);
        cvt_x_f16<<<(TOKENS * IN_F / 8) / 256, 256, 0, stream>>>(x, x16);
        deq_w_f16<<<dim3(OUT_F / 256, IN_F / 8), 256, 0, stream>>>(qweight, qzeros, scales, w16);
        gptq4_wmma_gemm_fast<<<grid, block, 0, stream>>>(x16, w16, bias, out);
    } else {
        gptq4_wmma_gemm_fused<<<grid, block, 0, stream>>>(x, qweight, qzeros, scales, bias, out);
    }
}
